// Decoder_66477503807624
// MI455X (gfx1250) — matrix operand order repaired and checked by reading, not compiled, not run on hardware
//
#include <hip/hip_runtime.h>
#include <hip/hip_bf16.h>

#define VSZ 30000
#define HSZ 512
#define BSZ 16
#define SSZ 256
#define JSZ 10
#define TSZ 8
#define NOPS 4
#define RSZ (JSZ*BSZ)     // 160 rows = (j,b) pairs
#define NEGV (-1e9f)

typedef __attribute__((ext_vector_type(8)))  float  v8f;
typedef __attribute__((ext_vector_type(2)))  float  v2f;
typedef __attribute__((ext_vector_type(16))) __bf16 v16bf;

union FragBF { v16bf v; unsigned int u[8]; };

__device__ __forceinline__ unsigned short f2bf(float x) {
  unsigned int u = __float_as_uint(x);
  u += 0x7fffu + ((u >> 16) & 1u);          // round-to-nearest-even
  return (unsigned short)(u >> 16);
}

// ---------------- one-time prep ----------------

__global__ void k_convert_emb(const float* __restrict__ src,
                              unsigned short* __restrict__ dst, int n) {
  int i = blockIdx.x * blockDim.x + threadIdx.x;
  if (i < n) dst[i] = f2bf(src[i]);
}

__global__ void k_init(const float* __restrict__ slot, const float* __restrict__ ench,
                       float* __restrict__ wbuf, float* __restrict__ hbuf) {
  int i = blockIdx.x * blockDim.x + threadIdx.x;
  if (i >= RSZ * HSZ) return;
  int r = i / HSZ, c = i % HSZ;
  int j = r / BSZ, b = r % BSZ;
  wbuf[i] = slot[j * HSZ + c];     // w0 = slot_emb broadcast over B
  hbuf[i] = ench[b * HSZ + c];     // h0 = encoder_hidden broadcast over J
}

// ---------------- GRU step: f32 WMMA 16x16x4 ----------------
// gi = w @ W_ih^T, gh = h @ W_hh^T; one wave per 16x16 tile of h_new.
// Needs 3 column-blocks (r,z,n gates) of gi and gh at the same tile -> 6 accumulators.

__global__ __launch_bounds__(32) void k_gru(
    const float* __restrict__ wbuf, const float* __restrict__ hin,
    const float* __restrict__ Wih, const float* __restrict__ Whh,
    const float* __restrict__ bih, const float* __restrict__ bhh,
    float* __restrict__ hout, unsigned short* __restrict__ hbf) {
  const int lane = threadIdx.x;
  const int nt = blockIdx.x & 31;        // 32 N-tiles over H=512
  const int mt = blockIdx.x >> 5;        // 10 M-tiles over R=160
  const int r0 = mt * 16, c0 = nt * 16;
  const int half = lane >> 4, lm = lane & 15;

  v8f z = {0.f,0.f,0.f,0.f,0.f,0.f,0.f,0.f};
  v8f aR=z, aZ=z, aN=z, hR=z, hZ=z, hN=z;

  const float* aw  = wbuf + (r0 + lm) * HSZ;
  const float* ah  = hin  + (r0 + lm) * HSZ;
  const float* bw0 = Wih + (size_t)(c0 +        lm) * HSZ;
  const float* bw1 = Wih + (size_t)(c0 +  512 + lm) * HSZ;
  const float* bw2 = Wih + (size_t)(c0 + 1024 + lm) * HSZ;
  const float* bh0 = Whh + (size_t)(c0 +        lm) * HSZ;
  const float* bh1 = Whh + (size_t)(c0 +  512 + lm) * HSZ;
  const float* bh2 = Whh + (size_t)(c0 + 1024 + lm) * HSZ;

  for (int k0 = 0; k0 < HSZ; k0 += 4) {
    int kb = k0 + half * 2;              // f32 A/B layout: lanes16-31 hold K=2,3
    v2f a_w = *reinterpret_cast<const v2f*>(aw  + kb);
    v2f a_h = *reinterpret_cast<const v2f*>(ah  + kb);
    v2f b0  = *reinterpret_cast<const v2f*>(bw0 + kb);
    v2f b1  = *reinterpret_cast<const v2f*>(bw1 + kb);
    v2f b2  = *reinterpret_cast<const v2f*>(bw2 + kb);
    v2f c0f = *reinterpret_cast<const v2f*>(bh0 + kb);
    v2f c1f = *reinterpret_cast<const v2f*>(bh1 + kb);
    v2f c2f = *reinterpret_cast<const v2f*>(bh2 + kb);
    aR = __builtin_amdgcn_wmma_f32_16x16x4_f32(false, a_w, false, b0,  (short)0, aR, false, false);
    aZ = __builtin_amdgcn_wmma_f32_16x16x4_f32(false, a_w, false, b1,  (short)0, aZ, false, false);
    aN = __builtin_amdgcn_wmma_f32_16x16x4_f32(false, a_w, false, b2,  (short)0, aN, false, false);
    hR = __builtin_amdgcn_wmma_f32_16x16x4_f32(false, a_h, false, c0f, (short)0, hR, false, false);
    hZ = __builtin_amdgcn_wmma_f32_16x16x4_f32(false, a_h, false, c1f, (short)0, hZ, false, false);
    hN = __builtin_amdgcn_wmma_f32_16x16x4_f32(false, a_h, false, c2f, (short)0, hN, false, false);
  }

#pragma unroll
  for (int i = 0; i < 8; ++i) {          // C/D layout: m = i + 8*half, n = lm
    int grow = r0 + i + half * 8;
    int c = c0 + lm;
    float ir  = aR[i] + bih[c];
    float iz  = aZ[i] + bih[512 + c];
    float inn = aN[i] + bih[1024 + c];
    float hr  = hR[i] + bhh[c];
    float hz  = hZ[i] + bhh[512 + c];
    float hn  = hN[i] + bhh[1024 + c];
    float rg = 1.f / (1.f + __expf(-(ir + hr)));
    float zg = 1.f / (1.f + __expf(-(iz + hz)));
    float ng = tanhf(inn + rg * hn);
    float hp = hin[grow * HSZ + c];
    float hv = (1.f - zg) * ng + zg * hp;
    hout[grow * HSZ + c] = hv;
    hbf[grow * HSZ + c]  = f2bf(hv);
  }
}

// ---------------- attention + context + p_gen (one block per row) ----------------

__global__ __launch_bounds__(256) void k_attn(
    const float* __restrict__ enc, const int* __restrict__ x,
    const float* __restrict__ h, const float* __restrict__ wbuf,
    const float* __restrict__ wgenW, const float* __restrict__ wgenb,
    float* __restrict__ attn, float* __restrict__ ctx, float* __restrict__ pgen) {
  __shared__ float red[256];
  __shared__ float aw[256];
  __shared__ float cbuf[512];
  int r = blockIdx.x, b = r % BSZ;
  int s = threadIdx.x;

  const float* er = enc + (size_t)(b * SSZ + s) * HSZ;
  const float* hr = h + (size_t)r * HSZ;
  float d = 0.f;
  for (int k = 0; k < HSZ; ++k) d += er[k] * hr[k];
  if (x[b * SSZ + s] == 0) d = NEGV;

  red[s] = d; __syncthreads();
  for (int off = 128; off > 0; off >>= 1) { if (s < off) red[s] = fmaxf(red[s], red[s + off]); __syncthreads(); }
  float m = red[0]; __syncthreads();
  float e = __expf(d - m);
  red[s] = e; __syncthreads();
  for (int off = 128; off > 0; off >>= 1) { if (s < off) red[s] += red[s + off]; __syncthreads(); }
  float a = e / red[0];
  aw[s] = a; attn[r * SSZ + s] = a; __syncthreads();

  for (int c = s; c < HSZ; c += 256) {
    float accv = 0.f;
    for (int s2 = 0; s2 < SSZ; ++s2) accv += aw[s2] * enc[(size_t)(b * SSZ + s2) * HSZ + c];
    cbuf[c] = accv; ctx[(size_t)r * HSZ + c] = accv;
  }
  __syncthreads();

  float p = 0.f;   // gen_in = [w, h, context] . w_gen_W
  for (int idx = s; idx < 3 * HSZ; idx += 256) {
    float g = (idx < HSZ) ? wbuf[r * HSZ + idx]
            : (idx < 2 * HSZ) ? h[r * HSZ + idx - HSZ]
            : cbuf[idx - 2 * HSZ];
    p += g * wgenW[idx];
  }
  red[s] = p; __syncthreads();
  for (int off = 128; off > 0; off >>= 1) { if (s < off) red[s] += red[s + off]; __syncthreads(); }
  if (s == 0) pgen[r] = 1.f / (1.f + __expf(-(red[0] + wgenb[0])));
}

// ---------------- vocab scores: bf16 WMMA 16x16x32, register-blocked ----------------
// Each wave computes a 32x80 output patch (2 M-tiles x 5 N-tiles, 10 accumulators):
// per k-step: 2 A-frag + 5 B-frag loads feed 10 WMMAs -> ~0.7KB L2 traffic per WMMA
// instead of 2KB with 1 tile/wave. 160/32=5 M-groups, 1875/5=375 N-groups, no tails.

#define VMB 2
#define VNB 5

__global__ __launch_bounds__(256) void k_vocab(
    const unsigned short* __restrict__ hbf, const unsigned short* __restrict__ ebf,
    float* __restrict__ outp, int t) {
  const int MG = RSZ / (16 * VMB);         // 5
  const int NG = VSZ / (16 * VNB);         // 375
  int wave = (blockIdx.x * 256 + threadIdx.x) >> 5;
  if (wave >= MG * NG) return;             // uniform per wave
  int lane = threadIdx.x & 31;
  int mg = wave % MG, ng = wave / MG;
  int r0 = mg * 16 * VMB, v0 = ng * 16 * VNB;
  int half = lane >> 4, lm = lane & 15;

  const unsigned short* arow0 = hbf + (size_t)(r0 + lm) * HSZ;
  const unsigned short* arow1 = arow0 + (size_t)16 * HSZ;
  const unsigned short* brow[VNB];
#pragma unroll
  for (int n = 0; n < VNB; ++n)
    brow[n] = ebf + (size_t)(v0 + n * 16 + lm) * HSZ;

  v8f zero = {0.f,0.f,0.f,0.f,0.f,0.f,0.f,0.f};
  v8f acc[VMB * VNB];
#pragma unroll
  for (int i = 0; i < VMB * VNB; ++i) acc[i] = zero;

  for (int k0 = 0; k0 < HSZ; k0 += 32) {
    FragBF a0, a1;
#pragma unroll
    for (int i = 0; i < 8; ++i) {          // 16-bit A 16x32 layout
      int ko = k0 + ((i >> 2) << 4) + (half << 3) + ((i & 3) << 1);
      a0.u[i] = *reinterpret_cast<const unsigned int*>(arow0 + ko);
      a1.u[i] = *reinterpret_cast<const unsigned int*>(arow1 + ko);
    }
#pragma unroll
    for (int n = 0; n < VNB; ++n) {
      FragBF bb;
#pragma unroll
      for (int i = 0; i < 8; ++i)          // B: same 16-bit 16x32 K layout as A
        bb.u[i] = *reinterpret_cast<const unsigned int*>(
            brow[n] + k0 + ((i >> 2) << 4) + (half << 3) + ((i & 3) << 1));
      acc[n * VMB + 0] = __builtin_amdgcn_wmma_f32_16x16x32_bf16(
          false, a0.v, false, bb.v, (short)0, acc[n * VMB + 0], false, false);
      acc[n * VMB + 1] = __builtin_amdgcn_wmma_f32_16x16x32_bf16(
          false, a1.v, false, bb.v, (short)0, acc[n * VMB + 1], false, false);
    }
  }

#pragma unroll
  for (int n = 0; n < VNB; ++n) {
#pragma unroll
    for (int mi = 0; mi < VMB; ++mi) {
#pragma unroll
      for (int i = 0; i < 8; ++i) {
        int rr = r0 + mi * 16 + i + half * 8;
        int jj = rr / BSZ, bb2 = rr % BSZ;
        outp[(size_t)((bb2 * JSZ + jj) * TSZ + t) * VSZ + v0 + n * 16 + lm] =
            acc[n * VMB + mi][i];
      }
    }
  }
}

// ---------------- finalize: softmax in place, pointer scatter, argmax, gather w_next ----------------

__global__ __launch_bounds__(256) void k_finalize(
    float* __restrict__ outp, const int* __restrict__ x,
    const float* __restrict__ attn, const float* __restrict__ pgen,
    const float* __restrict__ embW, float* __restrict__ wbuf, int t) {
  __shared__ float red[256];
  __shared__ int   redi[256];
  int r = blockIdx.x, j = r / BSZ, b = r % BSZ;
  int tid = threadIdx.x;
  float* base = outp + (size_t)((b * JSZ + j) * TSZ + t) * VSZ;

  float m = -3.4e38f;
  for (int v = tid; v < VSZ; v += 256) m = fmaxf(m, base[v]);
  red[tid] = m; __syncthreads();
  for (int off = 128; off > 0; off >>= 1) { if (tid < off) red[tid] = fmaxf(red[tid], red[tid + off]); __syncthreads(); }
  m = red[0]; __syncthreads();

  float sum = 0.f;
  for (int v = tid; v < VSZ; v += 256) sum += __expf(base[v] - m);
  red[tid] = sum; __syncthreads();
  for (int off = 128; off > 0; off >>= 1) { if (tid < off) red[tid] += red[tid + off]; __syncthreads(); }
  sum = red[0];

  float pg = pgen[r];
  float scale = pg / sum;
  for (int v = tid; v < VSZ; v += 256) base[v] = __expf(base[v] - m) * scale;
  __threadfence_block(); __syncthreads();

  if (tid == 0) {                        // deterministic serial scatter (pointer dist)
    float om = 1.f - pg;
    for (int s2 = 0; s2 < SSZ; ++s2) {
      int tok = x[b * SSZ + s2];
      if (tok != 0) base[tok] += om * attn[r * SSZ + s2];
    }
  }
  __threadfence_block(); __syncthreads();

  float bv = -3.4e38f; int bi = 0;       // argmax, first-index tie-break
  for (int v = tid; v < VSZ; v += 256) { float val = base[v]; if (val > bv) { bv = val; bi = v; } }
  red[tid] = bv; redi[tid] = bi; __syncthreads();
  for (int off = 128; off > 0; off >>= 1) {
    if (tid < off) {
      if (red[tid + off] > red[tid] ||
          (red[tid + off] == red[tid] && redi[tid + off] < redi[tid])) {
        red[tid] = red[tid + off]; redi[tid] = redi[tid + off];
      }
    }
    __syncthreads();
  }
  int widx = redi[0];
  for (int c = tid; c < HSZ; c += 256) wbuf[r * HSZ + c] = embW[(size_t)widx * HSZ + c];
}

// ---------------- gate head (t==0 context only) ----------------

__global__ __launch_bounds__(64) void k_gates(
    const float* __restrict__ ctx, const float* __restrict__ actW,
    const float* __restrict__ actb, float* __restrict__ outg) {
  __shared__ float red[64];
  int idx = blockIdx.x;
  int r = idx / NOPS, op = idx % NOPS;
  int j = r / BSZ, b = r % BSZ;
  int tid = threadIdx.x;
  float p = 0.f;
  for (int k = tid; k < HSZ; k += 64) p += ctx[r * HSZ + k] * actW[op * HSZ + k];
  red[tid] = p; __syncthreads();
  for (int off = 32; off > 0; off >>= 1) { if (tid < off) red[tid] += red[tid + off]; __syncthreads(); }
  if (tid == 0) outg[(b * JSZ + j) * NOPS + op] = red[0] + actb[op];
}

// ---------------- launcher ----------------

extern "C" void kernel_launch(void* const* d_in, const int* in_sizes, int n_in,
                              void* d_out, int out_size, void* d_ws, size_t ws_size,
                              hipStream_t stream) {
  (void)in_sizes; (void)n_in; (void)out_size; (void)ws_size;
  const int*   x     = (const int*)  d_in[0];
  const float* enc   = (const float*)d_in[1];
  const float* ench  = (const float*)d_in[2];
  // d_in[3] = max_len (device scalar) -- fixed T=8 per reference constants
  const float* embW  = (const float*)d_in[4];
  const float* slot  = (const float*)d_in[5];
  const float* Wih   = (const float*)d_in[6];
  const float* Whh   = (const float*)d_in[7];
  const float* bih   = (const float*)d_in[8];
  const float* bhh   = (const float*)d_in[9];
  const float* wgenW = (const float*)d_in[10];
  const float* wgenb = (const float*)d_in[11];
  const float* actW  = (const float*)d_in[12];
  const float* actb  = (const float*)d_in[13];

  float* outg = (float*)d_out;                 // (B,J,NOP) = 640 floats
  float* outp = outg + BSZ * JSZ * NOPS;       // (B,J,T,V)

  char* ws = (char*)d_ws;
  unsigned short* emb_bf = (unsigned short*)ws;
  size_t off = (size_t)VSZ * HSZ * 2;          // 30.72 MB, L2-resident weight copy
  float* wbuf = (float*)(ws + off); off += (size_t)RSZ * HSZ * 4;
  float* h0   = (float*)(ws + off); off += (size_t)RSZ * HSZ * 4;
  float* h1   = (float*)(ws + off); off += (size_t)RSZ * HSZ * 4;
  float* ctx  = (float*)(ws + off); off += (size_t)RSZ * HSZ * 4;
  float* attn = (float*)(ws + off); off += (size_t)RSZ * SSZ * 4;
  float* pgen = (float*)(ws + off); off += (size_t)RSZ * 4;
  unsigned short* hbf = (unsigned short*)(ws + off); off += (size_t)RSZ * HSZ * 2;

  k_convert_emb<<<(VSZ * HSZ + 255) / 256, 256, 0, stream>>>(embW, emb_bf, VSZ * HSZ);
  k_init<<<(RSZ * HSZ + 255) / 256, 256, 0, stream>>>(slot, ench, wbuf, h0);

  const int vocab_waves  = (RSZ / (16 * VMB)) * (VSZ / (16 * VNB));  // 5*375 = 1875
  const int vocab_blocks = (vocab_waves + 7) / 8;                    // 8 waves/block

  for (int t = 0; t < TSZ; ++t) {
    float* hin  = (t & 1) ? h1 : h0;
    float* hout = (t & 1) ? h0 : h1;
    k_gru<<<(RSZ / 16) * (HSZ / 16), 32, 0, stream>>>(wbuf, hin, Wih, Whh, bih, bhh, hout, hbf);
    k_attn<<<RSZ, 256, 0, stream>>>(enc, x, hout, wbuf, wgenW, wgenb, attn, ctx, pgen);
    if (t == 0)
      k_gates<<<RSZ * NOPS, 64, 0, stream>>>(ctx, actW, actb, outg);
    k_vocab<<<vocab_blocks, 256, 0, stream>>>(hbf, emb_bf, outp, t);
    k_finalize<<<RSZ, 256, 0, stream>>>(outp, x, attn, pgen, embW, wbuf, t);
  }
}
